// MultiHeadAttention2_3264175145360
// MI455X (gfx1250) — matrix operand order repaired and checked by reading, not compiled, not run on hardware
//
#include <hip/hip_runtime.h>
#include <hip/hip_bf16.h>
#include <type_traits>

typedef __bf16 bf16;
typedef __attribute__((ext_vector_type(16))) __bf16 v16bf;
typedef __attribute__((ext_vector_type(8)))  float  v8f;

#define BM 256   // block M tile (8 waves x 32 rows)
#define BN 64    // block N tile
#define BK 32    // K step
#define LDA 40   // bf16 pitch: 80B = 16B-aligned for async B128, 20-dword bank rotation
#define LDB 40

union FragBF { v16bf bf; unsigned u[8]; };
union Acc    { v8f v;    float f[8];    };

// CDNA5 async global->LDS copy (ASYNCcnt-tracked, bypasses VGPRs).
__device__ __forceinline__ void async_load_b128(unsigned lds_off, const void* gaddr) {
    asm volatile("global_load_async_to_lds_b128 %0, %1, off"
                 :: "v"(lds_off), "v"(gaddr) : "memory");
}
template <int N>
__device__ __forceinline__ void wait_async() {
    asm volatile("s_wait_asynccnt %0" :: "i"(N) : "memory");
}

// fp32 -> bf16 one-shot conversion (x as bf16 = 128MB fits MI455X 192MB L2)
__global__ __launch_bounds__(256)
void cvt_f32_bf16(const float* __restrict__ src, bf16* __restrict__ dst, long n4)
{
    long i = (long)blockIdx.x * 256 + threadIdx.x;
    if (i < n4) {
        float4 v = ((const float4*)src)[i];
        union { bf16 h[4]; uint2 u; } p;
        p.h[0] = (bf16)v.x; p.h[1] = (bf16)v.y;
        p.h[2] = (bf16)v.z; p.h[3] = (bf16)v.w;
        ((uint2*)dst)[i] = p.u;
    }
}

// Issue one stage of async copies: A tile BMxBK (4 chunks/thread),
// B tile BNxBK (1 chunk/thread) -> 5 ASYNCcnt ops per thread.
__device__ __forceinline__ void stage_tiles(const bf16* __restrict__ A,
                                            const bf16* __restrict__ Bw,
                                            long tileM, long tileN, int K, int kk,
                                            int tid, __bf16* bufA, __bf16* bufB)
{
    #pragma unroll
    for (int i = 0; i < 4; ++i) {
        int f   = tid + i * 256;
        int row = f >> 2;               // 4 x 16B chunks per 32-element row
        int m   = f & 3;
        const bf16* g = A + (tileM + row) * (long)K + kk + m * 8;
        async_load_b128((unsigned)(unsigned long long)&bufA[row * LDA + m * 8], g);
    }
    {
        int row = tid >> 2;
        int m   = tid & 3;
        const bf16* g = Bw + (tileN + row) * (long)K + kk + m * 8;
        async_load_b128((unsigned)(unsigned long long)&bufB[row * LDB + m * 8], g);
    }
}

// C[M,N] = A[M,K] @ Bw[N,K]^T + bias[N]   (torch Linear convention), bf16 inputs.
// Double-buffered async-LDS pipeline; each wave computes a 32x64 strip.
template <typename OT>
__global__ __launch_bounds__(256)
void gemm_bias_wmma(const bf16* __restrict__ A, const bf16* __restrict__ Bw,
                    const float* __restrict__ bias, OT* __restrict__ C,
                    int K, int ldc)
{
    __shared__ __align__(16) __bf16 lds_a[2][BM * LDA];
    __shared__ __align__(16) __bf16 lds_b[2][BN * LDB];

    const int tid   = threadIdx.x;
    const int lane  = tid & 31;
    const int wave  = tid >> 5;
    const long tileM = (long)blockIdx.y * BM;
    const long tileN = (long)blockIdx.x * BN;

    Acc c[2][4];
    #pragma unroll
    for (int h = 0; h < 2; ++h)
        #pragma unroll
        for (int t = 0; t < 4; ++t)
            #pragma unroll
            for (int i = 0; i < 8; ++i) c[h][t].f[i] = 0.0f;

    const int nsteps = K / BK;
    stage_tiles(A, Bw, tileM, tileN, K, 0, tid, lds_a[0], lds_b[0]);

    for (int s = 0; s < nsteps; ++s) {
        const int buf = s & 1;
        if (s + 1 < nsteps) {
            // prefetch next stage into the other buffer, then wait until only
            // those 5 copies remain in flight -> stage s has landed in LDS
            stage_tiles(A, Bw, tileM, tileN, K, (s + 1) * BK, tid,
                        lds_a[buf ^ 1], lds_b[buf ^ 1]);
            wait_async<5>();
        } else {
            wait_async<0>();
        }
        __syncthreads();      // all waves' stage-s data visible

        // ---- two A fragments (rows wave*32 .. +31), ISA 16-bit 16x32 layout ----
        FragBF a[2];
        #pragma unroll
        for (int h = 0; h < 2; ++h) {
            int lrow = wave * 32 + h * 16 + (lane & 15);
            const unsigned* arow = (const unsigned*)&lds_a[buf][lrow * LDA];
            int kb = (lane >= 16) ? 4 : 0;   // u32 units (K=8..15 / 24..31 halves)
            #pragma unroll
            for (int j = 0; j < 4; ++j) {
                a[h].u[j]     = arow[kb + j];      // K = kb*2 + 2j, +1
                a[h].u[4 + j] = arow[8 + kb + j];  // K = 16 + kb*2 + 2j, +1
            }
        }
        // ---- 4 B fragments, each reused by both A fragments: 8 WMMAs ----
        #pragma unroll
        for (int t = 0; t < 4; ++t) {
            FragBF b;
            int bcol = t * 16 + (lane & 15);
            const unsigned* brow = (const unsigned*)&lds_b[buf][bcol * LDB];
            int kb = (lane >= 16) ? 4 : 0;   // lanes 16-31 hold K=8..15 / 24..31
            #pragma unroll
            for (int j = 0; j < 4; ++j) { b.u[j] = brow[kb + j]; b.u[4 + j] = brow[8 + kb + j]; }
            c[0][t].v = __builtin_amdgcn_wmma_f32_16x16x32_bf16(
                false, a[0].bf, false, b.bf, (short)0, c[0][t].v, false, false);
            c[1][t].v = __builtin_amdgcn_wmma_f32_16x16x32_bf16(
                false, a[1].bf, false, b.bf, (short)0, c[1][t].v, false, false);
        }
        __syncthreads();      // done reading buf before it is overwritten
    }

    // ---- epilogue: C layout (VGPR i -> M=i / M=8+i), fused bias ----
    #pragma unroll
    for (int h = 0; h < 2; ++h) {
        #pragma unroll
        for (int t = 0; t < 4; ++t) {
            #pragma unroll
            for (int i = 0; i < 8; ++i) {
                int  m   = (lane < 16) ? i : (8 + i);
                int  n   = lane & 15;
                long row = tileM + wave * 32 + h * 16 + m;
                long col = tileN + t * 16 + n;
                float val = c[h][t].f[i] + bias[col];
                if constexpr (std::is_same<OT, float>::value)
                    C[row * (long)ldc + col] = val;
                else
                    C[row * (long)ldc + col] = (__bf16)val;
            }
        }
    }
}

// N=2 attention: one wave per (b,h). 2x2 logits via wave32 reductions,
// softmax, V-combine, and mean over N (mean commutes with the projection).
__global__ __launch_bounds__(256)
void attn_n2(const bf16* __restrict__ qkv, bf16* __restrict__ avg)
{
    const int  lane = threadIdx.x & 31;
    const int  wave = threadIdx.x >> 5;
    const long gid  = (long)blockIdx.x * 8 + wave;   // = b*16 + h
    const long b    = gid >> 4;
    const int  h    = (int)(gid & 15);

    const bf16* r0 = qkv + (b * 2 + 0) * 3072 + h * 64;  // q|k|v at +0,+1024,+2048
    const bf16* r1 = qkv + (b * 2 + 1) * 3072 + h * 64;

    const int d0 = lane, d1 = lane + 32;
    float q0a = (float)r0[d0],        q0b = (float)r0[d1];
    float q1a = (float)r1[d0],        q1b = (float)r1[d1];
    float k0a = (float)r0[1024 + d0], k0b = (float)r0[1024 + d1];
    float k1a = (float)r1[1024 + d0], k1b = (float)r1[1024 + d1];
    float v0a = (float)r0[2048 + d0], v0b = (float)r0[2048 + d1];
    float v1a = (float)r1[2048 + d0], v1b = (float)r1[2048 + d1];

    float s00 = q0a * k0a + q0b * k0b;
    float s01 = q0a * k1a + q0b * k1b;
    float s10 = q1a * k0a + q1b * k0b;
    float s11 = q1a * k1a + q1b * k1b;
    #pragma unroll
    for (int off = 16; off; off >>= 1) {
        s00 += __shfl_xor(s00, off, 32);
        s01 += __shfl_xor(s01, off, 32);
        s10 += __shfl_xor(s10, off, 32);
        s11 += __shfl_xor(s11, off, 32);
    }
    const float scale = 0.125f;                      // 64^-0.5
    s00 *= scale; s01 *= scale; s10 *= scale; s11 *= scale;
    float m0  = fmaxf(s00, s01), m1 = fmaxf(s10, s11);
    float e00 = __expf(s00 - m0), e01 = __expf(s01 - m0);
    float e10 = __expf(s10 - m1), e11 = __expf(s11 - m1);
    float i0  = 1.0f / (e00 + e01), i1 = 1.0f / (e10 + e11);
    float c0 = 0.5f * (e00 * i0 + e10 * i1);
    float c1 = 0.5f * (e01 * i0 + e11 * i1);

    bf16* out = avg + b * 1024 + h * 64;
    out[d0] = (bf16)(c0 * v0a + c1 * v1a);
    out[d1] = (bf16)(c0 * v0b + c1 * v1b);
}

extern "C" void kernel_launch(void* const* d_in, const int* in_sizes, int n_in,
                              void* d_out, int out_size, void* d_ws, size_t ws_size,
                              hipStream_t stream) {
    const float* x      = (const float*)d_in[0];  // [32768, 2, 1024]
    const float* w_qkv  = (const float*)d_in[1];  // [3072, 1024]
    const float* b_qkv  = (const float*)d_in[2];  // [3072]
    const float* w_proj = (const float*)d_in[3];  // [1024, 1024]
    const float* b_proj = (const float*)d_in[4];  // [1024]
    float* out = (float*)d_out;                   // [32768, 1024]

    const long B = 32768, N = 2, D = 1024;
    const long nX = B * N * D;          // 67,108,864
    const long nWq = 3 * D * D;         // 3,145,728
    const long nWp = D * D;             // 1,048,576

    bf16* xbf   = (bf16*)d_ws;
    bf16* wqkvb = xbf + nX;
    bf16* wprjb = wqkvb + nWq;
    bf16* qkv   = wprjb + nWp;          // [B*N, 3D]
    bf16* avg   = qkv + B * N * 3 * D;  // [B, D]

    // 0) one-shot fp32 -> bf16
    cvt_f32_bf16<<<(unsigned)(nX / 4 / 256), 256, 0, stream>>>(x, xbf, nX / 4);
    cvt_f32_bf16<<<(unsigned)(nWq / 4 / 256), 256, 0, stream>>>(w_qkv, wqkvb, nWq / 4);
    cvt_f32_bf16<<<(unsigned)(nWp / 4 / 256), 256, 0, stream>>>(w_proj, wprjb, nWp / 4);

    // 1) fused QKV projection: 65536x3072 = x @ w_qkv^T + b_qkv
    gemm_bias_wmma<bf16>
        <<<dim3((unsigned)(3 * D / BN), (unsigned)(B * N / BM)), 256, 0, stream>>>(
            xbf, wqkvb, b_qkv, qkv, (int)D, (int)(3 * D));

    // 2) N=2 attention + mean over N (one wave per (b,h))
    attn_n2<<<(unsigned)(B * 16 / 8), 256, 0, stream>>>(qkv, avg);

    // 3) output projection on the averaged rows: 32768x1024
    gemm_bias_wmma<float>
        <<<dim3((unsigned)(D / BN), (unsigned)(B / BM)), 256, 0, stream>>>(
            avg, wprjb, b_proj, out, (int)D, (int)D);
}
